// TransformerEncoder_4071628996638
// MI455X (gfx1250) — matrix operand order repaired and checked by reading, not compiled, not run on hardware
//
#include <hip/hip_runtime.h>
#include <hip/hip_bf16.h>
#include <math.h>
#include <cstdint>

// ---------------------------------------------------------------------------
// MI455X (gfx1250) transformer encoder layer.
// bf16 WMMA (v_wmma_f32_16x16x32_bf16) + f32 accumulate; TDM (tensor_load_to_lds)
// stages all GEMM tiles asynchronously; flash attention avoids ~400MB of
// score-matrix HBM traffic; softmax reductions use v_permlane16_b32 (VALU).
// ---------------------------------------------------------------------------

typedef __bf16 bf16;
typedef __attribute__((ext_vector_type(16))) __bf16 v16bf;
typedef __attribute__((ext_vector_type(8)))  float  v8f;
typedef __attribute__((ext_vector_type(4)))  unsigned int v4u;
typedef __attribute__((ext_vector_type(8)))  int v8i;
typedef __attribute__((ext_vector_type(4)))  int v4i;

#define FEATS 768
#define NHEAD 12
#define DHEAD 64
#define SEQ   1024
#define BATCH 8
#define ROWS  (BATCH * SEQ)   /* 8192 */
#define HID   3072

#define BM 128
#define BN 128
#define BK 32

union AB16 { v16bf v; uint4 q[2]; };
union TV16 { uint4 q[2]; bf16 e[16]; };

// ---------------------------------------------------------------------------
// TDM: 2D tile load, global -> LDS.  D# packing per CDNA5 ISA
// §8.3/§8.4: group0 = {count=1, lds_addr, global_addr[56:0], type=2};
// group1 = {mask=0, data_size=1(2B), dims, tile dims, stride0}; groups 2/3 = 0.
// ---------------------------------------------------------------------------
__device__ __forceinline__ void tdm_load_2d(unsigned lds_addr, const void* gptr,
                                            unsigned tensor_d0, unsigned tensor_d1,
                                            unsigned tile_d0, unsigned tile_d1,
                                            unsigned long long stride0_elems) {
  const unsigned long long ga = (unsigned long long)(uintptr_t)gptr;
  v4u g0;
  g0.x = 1u;                                   // count=1, user descriptor
  g0.y = lds_addr;                             // LDS byte address
  g0.z = (unsigned)(ga & 0xffffffffu);         // global_addr[31:0]
  g0.w = (unsigned)((ga >> 32) & 0x01ffffffu) | (2u << 30);  // ga[56:32], type=2
  v8i g1;
  g1[0] = (int)(1u << 16);                     // workgroup_mask=0, data_size=1 (2B)
  g1[1] = (int)((tensor_d0 & 0xffffu) << 16);  // tensor_dim0[15:0]
  g1[2] = (int)(((tensor_d0 >> 16) & 0xffffu) | ((tensor_d1 & 0xffffu) << 16));
  g1[3] = (int)(((tensor_d1 >> 16) & 0xffffu) | ((tile_d0 & 0xffffu) << 16));
  g1[4] = (int)(tile_d1 & 0xffffu);            // tile_dim1; tile_dim2 = 0
  g1[5] = (int)(unsigned)(stride0_elems & 0xffffffffu);       // dim0_stride[31:0]
  g1[6] = (int)(unsigned)((stride0_elems >> 32) & 0xffffu);   // [47:32]; dim1_stride=0
  g1[7] = 0;
  v4i z4 = {0, 0, 0, 0};
#if __has_include(<hip/amd_detail/amd_gfx1250_TDM.h>)
  v8i z8 = {0, 0, 0, 0, 0, 0, 0, 0};           // therock 6-arg variant
  __builtin_amdgcn_tensor_load_to_lds(g0, g1, z4, z4, z8, 0);
#else
  __builtin_amdgcn_tensor_load_to_lds(g0, g1, z4, z4, 0);    // ROCm 7.2 5-arg
#endif
}

__device__ __forceinline__ unsigned lds_byte_addr(const void* p) {
  // flat shared addresses carry the LDS byte offset in addr[31:0] (ISA §10.2)
  return (unsigned)(uintptr_t)p;
}

// ---------------------------------------------------------------------------
// WMMA fragment helpers (ISA 7.12.2 layouts)
// ---------------------------------------------------------------------------
__device__ __forceinline__ v16bf a_frag_lds(const bf16* s, int row0, int pitch, int kofs) {
  const int lane = threadIdx.x & 31;
  const int m = lane & 15, hf = lane >> 4;
  const bf16* p = s + (size_t)(row0 + m) * pitch + kofs + hf * 8;
  AB16 u;
  u.q[0] = *(const uint4*)(p);
  u.q[1] = *(const uint4*)(p + 16);
  return u.v;
}

__device__ __forceinline__ v16bf b_frag_lds(const bf16* s, int col0, int pitch, int kofs) {
  const int lane = threadIdx.x & 31;
  const int n = lane & 15, hf = lane >> 4;
  const bf16* p = s + (size_t)(col0 + n) * pitch + kofs + hf * 8;
  AB16 u;
  u.q[0] = *(const uint4*)(p);
  u.q[1] = *(const uint4*)(p + 16);
  return u.v;
}

__device__ __forceinline__ v8f wmma_bf16(v16bf a, v16bf b, v8f c) {
  return __builtin_amdgcn_wmma_f32_16x16x32_bf16(false, a, false, b, (short)0, c,
                                                 false, false);
}

// XOR-butterfly within 16-lane rows via v_permlane16_b32 (pure VALU, no LDS)
__device__ __forceinline__ float permxor16(float v, unsigned s0, unsigned s1) {
  unsigned u = __builtin_bit_cast(unsigned, v);
  u = __builtin_amdgcn_permlane16(u, u, s0, s1, false, false);
  return __builtin_bit_cast(float, u);
}
#define XOR1_LO 0x67452301u
#define XOR1_HI 0xEFCDAB89u
#define XOR2_LO 0x54761032u
#define XOR2_HI 0xDCFE98BAu
#define XOR4_LO 0x32107654u
#define XOR4_HI 0xBA98FEDCu
#define XOR8_LO 0xFEDCBA98u
#define XOR8_HI 0x76543210u

// ---------------------------------------------------------------------------
// fp32 -> bf16 convert
// ---------------------------------------------------------------------------
__global__ void f32_to_bf16_kernel(const float* __restrict__ in,
                                   bf16* __restrict__ out, int n) {
  int i = (blockIdx.x * blockDim.x + threadIdx.x) * 8;
  if (i + 8 <= n) {
    float4 a = ((const float4*)(in + i))[0];
    float4 b = ((const float4*)(in + i))[1];
    TV16 t;
    t.e[0] = (bf16)a.x; t.e[1] = (bf16)a.y; t.e[2] = (bf16)a.z; t.e[3] = (bf16)a.w;
    t.e[4] = (bf16)b.x; t.e[5] = (bf16)b.y; t.e[6] = (bf16)b.z; t.e[7] = (bf16)b.w;
    *(uint4*)(out + i) = t.q[0];
  }
}

// ---------------------------------------------------------------------------
// Weight pre-pass: W[K][N] f32 -> WT[N][K] bf16 (one-time, ~14MB total).
// Removes all per-stage scatter stores from the hot GEMM loop.
// ---------------------------------------------------------------------------
__global__ void transpose_bf16_kernel(const float* __restrict__ W,
                                      bf16* __restrict__ WT, int K, int N) {
  __shared__ bf16 tile[32][33];
  const int nb = blockIdx.x % (N / 32);
  const int kb = blockIdx.x / (N / 32);
  const int tx = threadIdx.x & 31;
  const int ty = threadIdx.x >> 5;  // 0..7
#pragma unroll
  for (int i = 0; i < 4; ++i) {
    const int k = kb * 32 + ty + i * 8;
    tile[tx][ty + i * 8] = (bf16)W[(size_t)k * N + nb * 32 + tx];
  }
  __syncthreads();
#pragma unroll
  for (int i = 0; i < 4; ++i) {
    const int n = nb * 32 + ty + i * 8;
    WT[(size_t)n * K + kb * 32 + tx] = tile[ty + i * 8][tx];
  }
}

// ---------------------------------------------------------------------------
// WMMA GEMM: C[M,N] = A[M,K](bf16) x WT[N,K](bf16). Block tile 128x128, k=32.
// Both tiles staged by TDM (tensor_load_to_lds), double buffered; wave 0 issues
// descriptors and gates the barrier with s_wait_tensorcnt.
// ---------------------------------------------------------------------------
#define EPI_BIAS_BF16 0
#define EPI_RES       1
#define EPI_GELU      2
#define EPI_GELU_RES  3

template <int EPI>
__global__ __launch_bounds__(256, 2) void gemm_wmma_kernel(
    const bf16* __restrict__ A, const bf16* __restrict__ WT,
    const float* __restrict__ bias, const float* __restrict__ resid,
    float* __restrict__ outF, bf16* __restrict__ outB, int M, int N, int K) {
  __shared__ bf16 sA[2][BM * BK];   // [m][k]
  __shared__ bf16 sB[2][BN * BK];   // [n][k]

  const int nblocks = N / BN;
  const int mblk = blockIdx.x / nblocks;
  const int nblk = blockIdx.x % nblocks;
  const int t = threadIdx.x;
  const int wid = t >> 5;
  const int wm = wid >> 1;  // 0..3
  const int wn = wid & 1;   // 0..1
  const int lane = t & 31;
  const int ln = lane & 15, hf = lane >> 4;

  v8f acc[2][4];
#pragma unroll
  for (int mt = 0; mt < 2; ++mt)
#pragma unroll
    for (int nt = 0; nt < 4; ++nt) acc[mt][nt] = (v8f)0.0f;

  auto issue_stage = [&](int k0, int buf) {
    tdm_load_2d(lds_byte_addr(&sA[buf][0]), A + (size_t)mblk * BM * K + k0,
                (unsigned)K, (unsigned)M, BK, BM, (unsigned long long)K);
    tdm_load_2d(lds_byte_addr(&sB[buf][0]), WT + (size_t)nblk * BN * K + k0,
                (unsigned)K, (unsigned)N, BK, BN, (unsigned long long)K);
  };

  if (wid == 0) {
    issue_stage(0, 0);
    __builtin_amdgcn_s_wait_tensorcnt(0);
  }
  __syncthreads();

  const int nk = K / BK;
  for (int ks = 0; ks < nk; ++ks) {
    const int buf = ks & 1;
    if (wid == 0 && ks + 1 < nk) issue_stage((ks + 1) * BK, buf ^ 1);

    v16bf aF[2], bF[4];
    aF[0] = a_frag_lds(&sA[buf][0], wm * 32, BK, 0);
    aF[1] = a_frag_lds(&sA[buf][0], wm * 32 + 16, BK, 0);
#pragma unroll
    for (int nt = 0; nt < 4; ++nt)
      bF[nt] = b_frag_lds(&sB[buf][0], wn * 64 + nt * 16, BK, 0);
#pragma unroll
    for (int mt = 0; mt < 2; ++mt)
#pragma unroll
      for (int nt = 0; nt < 4; ++nt)
        acc[mt][nt] = wmma_bf16(aF[mt], bF[nt], acc[mt][nt]);

    if (wid == 0 && ks + 1 < nk) __builtin_amdgcn_s_wait_tensorcnt(0);
    __syncthreads();
  }

  // Epilogue (C layout: VGPR r -> rows r / r+8, lanes -> cols)
  const int rowbase = mblk * BM + wm * 32;
  const int colbase = nblk * BN + wn * 64;
#pragma unroll
  for (int mt = 0; mt < 2; ++mt) {
#pragma unroll
    for (int nt = 0; nt < 4; ++nt) {
#pragma unroll
      for (int r = 0; r < 8; ++r) {
        const int row = rowbase + mt * 16 + r + hf * 8;
        const int col = colbase + nt * 16 + ln;
        float v = acc[mt][nt][r] + bias[col];
        if (EPI == EPI_GELU || EPI == EPI_GELU_RES)
          v = 0.5f * v * (1.0f + erff(v * 0.70710678118654752f));  // exact GELU
        if (EPI == EPI_RES || EPI == EPI_GELU_RES)
          v += resid[(size_t)row * N + col];
        if (EPI == EPI_RES || EPI == EPI_GELU_RES)
          outF[(size_t)row * N + col] = v;
        if (EPI == EPI_BIAS_BF16 || EPI == EPI_RES || EPI == EPI_GELU)
          outB[(size_t)row * N + col] = (bf16)v;
      }
    }
  }
}

// ---------------------------------------------------------------------------
// Flash attention: grid = (8 row-blocks) x (batch x heads).
// K tile staged by TDM; V tile loaded+transposed by all threads; softmax
// reductions via v_permlane16_b32; scale = 1/sqrt(FEATS) per the reference.
// ---------------------------------------------------------------------------
__global__ __launch_bounds__(256, 2) void attn_wmma_kernel(
    const bf16* __restrict__ Q, const bf16* __restrict__ Kmat,
    const bf16* __restrict__ V, bf16* __restrict__ O) {
  __shared__ bf16 sK[64 * 64];       // [key][d]  (B-frags for Q.K^T)
  __shared__ bf16 sVt[64 * 64];      // [d][key]  (B-frags for P.V)
  __shared__ bf16 sP[8][16 * 64];    // per-wave P transpose scratch

  const int rb = blockIdx.x & 7;
  const int bh = blockIdx.x >> 3;
  const int b = bh / NHEAD, h = bh % NHEAD;
  const int t = threadIdx.x, wid = t >> 5, lane = t & 31;
  const int ln = lane & 15, hf = lane >> 4;
  const int hofs = h * DHEAD;
  const size_t qrow0 = (size_t)b * SEQ + rb * 128 + wid * 16;

  // Q fragments (A layout, k = d_head, two 32-k steps), loaded once
  v16bf qF[2];
  {
    const bf16* p = Q + (qrow0 + ln) * FEATS + hofs;
    AB16 u;
    u.q[0] = *(const uint4*)(p + hf * 8);
    u.q[1] = *(const uint4*)(p + hf * 8 + 16);
    qF[0] = u.v;
    u.q[0] = *(const uint4*)(p + 32 + hf * 8);
    u.q[1] = *(const uint4*)(p + 32 + hf * 8 + 16);
    qF[1] = u.v;
  }

  v8f oacc[4];
#pragma unroll
  for (int nt = 0; nt < 4; ++nt) oacc[nt] = (v8f)0.0f;
  float mrow[8], lrow[8];
#pragma unroll
  for (int r = 0; r < 8; ++r) { mrow[r] = -1e30f; lrow[r] = 0.0f; }

  const float scale = 0.03608439182435161f;  // 1/sqrt(768)

  for (int jb = 0; jb < SEQ / 64; ++jb) {
    __syncthreads();  // previous tile fully consumed
    if (wid == 0) {   // K tile via TDM: 64 keys x 64 d, row stride FEATS
      tdm_load_2d(lds_byte_addr(&sK[0]),
                  Kmat + ((size_t)b * SEQ + jb * 64) * FEATS + hofs,
                  FEATS, ROWS, 64, 64, FEATS);
    }
    {  // V tile: load + transpose to [d][key]
      const int row = t >> 2;
      const int c0 = (t & 3) * 16;
      const size_t grow = ((size_t)b * SEQ + jb * 64 + row) * FEATS + hofs + c0;
      TV16 tv;
      tv.q[0] = *(const uint4*)(V + grow);
      tv.q[1] = *(const uint4*)(V + grow + 8);
#pragma unroll
      for (int i = 0; i < 16; ++i) sVt[(c0 + i) * 64 + row] = tv.e[i];
      if (jb + 1 < SEQ / 64)  // global_prefetch_b8 of next V tile
        __builtin_prefetch(V + grow + (size_t)64 * FEATS, 0, 1);
    }
    if (wid == 0) __builtin_amdgcn_s_wait_tensorcnt(0);
    __syncthreads();

    // S = Q K^T  (16 x 64 per wave)
    v8f sacc[4];
#pragma unroll
    for (int nt = 0; nt < 4; ++nt) sacc[nt] = (v8f)0.0f;
#pragma unroll
    for (int kst = 0; kst < 2; ++kst) {
#pragma unroll
      for (int nt = 0; nt < 4; ++nt) {
        v16bf bK = b_frag_lds(sK, nt * 16, 64, kst * 32);
        sacc[nt] = wmma_bf16(qF[kst], bK, sacc[nt]);
      }
    }

    // online softmax (VALU-only cross-lane reduction)
#pragma unroll
    for (int r = 0; r < 8; ++r) {
      float mx = fmaxf(fmaxf(sacc[0][r] * scale, sacc[1][r] * scale),
                       fmaxf(sacc[2][r] * scale, sacc[3][r] * scale));
      mx = fmaxf(mx, permxor16(mx, XOR1_LO, XOR1_HI));
      mx = fmaxf(mx, permxor16(mx, XOR2_LO, XOR2_HI));
      mx = fmaxf(mx, permxor16(mx, XOR4_LO, XOR4_HI));
      mx = fmaxf(mx, permxor16(mx, XOR8_LO, XOR8_HI));
      const float mnew = fmaxf(mrow[r], mx);
      const float alpha = __expf(mrow[r] - mnew);
      mrow[r] = mnew;
      float s = 0.0f;
#pragma unroll
      for (int nt = 0; nt < 4; ++nt) {
        float p = __expf(sacc[nt][r] * scale - mnew);
        sacc[nt][r] = p;
        s += p;
      }
      s += permxor16(s, XOR1_LO, XOR1_HI);
      s += permxor16(s, XOR2_LO, XOR2_HI);
      s += permxor16(s, XOR4_LO, XOR4_HI);
      s += permxor16(s, XOR8_LO, XOR8_HI);
      lrow[r] = lrow[r] * alpha + s;
#pragma unroll
      for (int nt = 0; nt < 4; ++nt) oacc[nt][r] *= alpha;
    }

    // P: C-layout -> A-layout via wave-private LDS (DS in-order; pin with wait)
    bf16* myP = &sP[wid][0];
#pragma unroll
    for (int nt = 0; nt < 4; ++nt)
#pragma unroll
      for (int r = 0; r < 8; ++r)
        myP[(r + hf * 8) * 64 + nt * 16 + ln] = (bf16)sacc[nt][r];
    asm volatile("s_wait_dscnt 0" ::: "memory");

    // O += P V
#pragma unroll
    for (int kst = 0; kst < 2; ++kst) {
      v16bf pF = a_frag_lds(myP, 0, 64, kst * 32);
#pragma unroll
      for (int nt = 0; nt < 4; ++nt) {
        v16bf bV = b_frag_lds(sVt, nt * 16, 64, kst * 32);
        oacc[nt] = wmma_bf16(pF, bV, oacc[nt]);
      }
    }
  }

  // normalize and store [b, i, h*64+d] as bf16 (feeds the Wo GEMM)
#pragma unroll
  for (int nt = 0; nt < 4; ++nt) {
#pragma unroll
    for (int r = 0; r < 8; ++r) {
      const float v = oacc[nt][r] / lrow[r];
      const size_t row = qrow0 + r + hf * 8;
      O[row * FEATS + hofs + nt * 16 + ln] = (bf16)v;
    }
  }
}

// ---------------------------------------------------------------------------
extern "C" void kernel_launch(void* const* d_in, const int* in_sizes, int n_in,
                              void* d_out, int out_size, void* d_ws,
                              size_t ws_size, hipStream_t stream) {
  (void)in_sizes; (void)n_in; (void)out_size; (void)ws_size;
  const float* x  = (const float*)d_in[0];
  const float* wq = (const float*)d_in[1];
  const float* bq = (const float*)d_in[2];
  const float* wk = (const float*)d_in[3];
  const float* bk = (const float*)d_in[4];
  const float* wv = (const float*)d_in[5];
  const float* bv = (const float*)d_in[6];
  const float* wo = (const float*)d_in[7];
  const float* bo = (const float*)d_in[8];
  const float* w1 = (const float*)d_in[9];
  const float* b1 = (const float*)d_in[10];
  const float* w2 = (const float*)d_in[11];
  const float* b2 = (const float*)d_in[12];
  float* out = (float*)d_out;

  char* ws = (char*)d_ws;
  size_t off = 0;
  auto alloc = [&](size_t bytes) -> char* {
    char* p = ws + off;
    off += (bytes + 255) & ~(size_t)255;
    return p;
  };
  const size_t EF = (size_t)ROWS * FEATS;
  const size_t WF = (size_t)FEATS * FEATS;
  const size_t WH = (size_t)FEATS * HID;
  bf16* xb    = (bf16*)alloc(EF * 2);
  bf16* qb    = (bf16*)alloc(EF * 2);
  bf16* kb    = (bf16*)alloc(EF * 2);
  bf16* vb    = (bf16*)alloc(EF * 2);
  bf16* attnb = (bf16*)alloc(EF * 2);
  float* out1f = (float*)alloc(EF * 4);
  bf16* hdnb  = (bf16*)alloc((size_t)ROWS * HID * 2);
  bf16* wqT = (bf16*)alloc(WF * 2);
  bf16* wkT = (bf16*)alloc(WF * 2);
  bf16* wvT = (bf16*)alloc(WF * 2);
  bf16* woT = (bf16*)alloc(WF * 2);
  bf16* w1T = (bf16*)alloc(WH * 2);
  bf16* w2T = (bf16*)alloc(WH * 2);
  bf16* out1b = xb;  // xb dead after QKV projections; reuse for residual-1 bf16

  // 0) one-time weight transpose+convert: W[K][N] f32 -> WT[N][K] bf16
  const int gW  = (FEATS / 32) * (FEATS / 32);  // 576
  const int gW1 = (FEATS / 32) * (HID / 32);    // 2304
  transpose_bf16_kernel<<<gW, 256, 0, stream>>>(wq, wqT, FEATS, FEATS);
  transpose_bf16_kernel<<<gW, 256, 0, stream>>>(wk, wkT, FEATS, FEATS);
  transpose_bf16_kernel<<<gW, 256, 0, stream>>>(wv, wvT, FEATS, FEATS);
  transpose_bf16_kernel<<<gW, 256, 0, stream>>>(wo, woT, FEATS, FEATS);
  transpose_bf16_kernel<<<gW1, 256, 0, stream>>>(w1, w1T, FEATS, HID);
  transpose_bf16_kernel<<<gW1, 256, 0, stream>>>(w2, w2T, HID, FEATS);

  // 1) x -> bf16
  f32_to_bf16_kernel<<<(int)(EF / (256 * 8)), 256, 0, stream>>>(x, xb, (int)EF);

  // 2) QKV projections
  const int gFF = (ROWS / BM) * (FEATS / BN);  // 384
  gemm_wmma_kernel<EPI_BIAS_BF16><<<gFF, 256, 0, stream>>>(
      xb, wqT, bq, nullptr, nullptr, qb, ROWS, FEATS, FEATS);
  gemm_wmma_kernel<EPI_BIAS_BF16><<<gFF, 256, 0, stream>>>(
      xb, wkT, bk, nullptr, nullptr, kb, ROWS, FEATS, FEATS);
  gemm_wmma_kernel<EPI_BIAS_BF16><<<gFF, 256, 0, stream>>>(
      xb, wvT, bv, nullptr, nullptr, vb, ROWS, FEATS, FEATS);

  // 3) flash attention
  attn_wmma_kernel<<<8 * BATCH * NHEAD, 256, 0, stream>>>(qb, kb, vb, attnb);

  // 4) out projection + residual (f32 for later residual, bf16 for MLP input)
  gemm_wmma_kernel<EPI_RES><<<gFF, 256, 0, stream>>>(
      attnb, woT, bo, x, out1f, out1b, ROWS, FEATS, FEATS);

  // 5) MLP-1: GELU(out1 @ w1 + b1) -> bf16 hidden
  const int gFH = (ROWS / BM) * (HID / BN);  // 1536
  gemm_wmma_kernel<EPI_GELU><<<gFH, 256, 0, stream>>>(
      out1b, w1T, b1, nullptr, nullptr, hdnb, ROWS, HID, FEATS);

  // 6) MLP-2: GELU(hdn @ w2 + b2) + out1 -> final f32 output
  gemm_wmma_kernel<EPI_GELU_RES><<<gFF, 256, 0, stream>>>(
      hdnb, w2T, b2, out1f, out, nullptr, ROWS, FEATS, HID);
}
